// CrossAttention_32770600469027
// MI455X (gfx1250) — matrix operand order repaired and checked by reading, not compiled, not run on hardware
//
#include <hip/hip_runtime.h>
#include <hip/hip_bf16.h>

typedef _Float16 v16h __attribute__((ext_vector_type(16)));
typedef _Float16 h8   __attribute__((ext_vector_type(8)));
typedef _Float16 h2   __attribute__((ext_vector_type(2)));
typedef float    v8f  __attribute__((ext_vector_type(8)));
typedef unsigned int u32x4 __attribute__((ext_vector_type(4)));
typedef int          i32x4 __attribute__((ext_vector_type(4)));
typedef int          i32x8 __attribute__((ext_vector_type(8)));

#define WMMA16x16x32(a, b, c) \
  __builtin_amdgcn_wmma_f32_16x16x32_f16(false, (a), false, (b), (short)0, (c), false, false)

// Problem constants
#define BATCH 4
#define LQ    2048
#define LK    2048
#define DIM   768
#define NH    12
#define HD    64

// --- Tensor Data Mover availability (device pass only; host pass falls back) ---
#if defined(__AMDGCN__) && __has_builtin(__builtin_amdgcn_tensor_load_to_lds) && \
    __has_builtin(__builtin_amdgcn_s_wait_tensorcnt)
#define HAVE_TDM 1
#else
#define HAVE_TDM 0
#endif

__device__ __forceinline__ unsigned int lds_lo32(const void* p) {
  return (unsigned int)(uintptr_t)p;  // generic LDS ptr: offset lives in [31:0]
}

#if HAVE_TDM
// 2-D f16 tile load via TDM: tile1 rows x tile0 halves, row stride stride0 halves,
// LDS padding: 1 DWORD after every (2<<pad_int) DWORDs stored.
__device__ __forceinline__ void tdm_load_tile16(unsigned int lds_addr, const void* gsrc,
                                                unsigned int dim0, unsigned int dim1,
                                                unsigned int tile0, unsigned int tile1,
                                                unsigned int stride0, unsigned int pad_int) {
  const unsigned long long ga = (unsigned long long)(uintptr_t)gsrc;
  u32x4 g0;
  g0[0] = 1u;                                   // count=1, user descriptor
  g0[1] = lds_addr;                             // bits[63:32]
  g0[2] = (unsigned int)ga;                     // global_addr[31:0]
  g0[3] = ((unsigned int)(ga >> 32) & 0x01FFFFFFu) | 0x80000000u;  // addr[56:32] | type=2
  i32x8 g1;
  g1[0] = (int)((1u << 16) | (1u << 20) | (pad_int << 22));  // data=2B, pad_en, pad=1 DWORD
  g1[1] = (int)(dim0 << 16);                                 // tensor_dim0[15:0]
  g1[2] = (int)((dim0 >> 16) | (dim1 << 16));                // dim0[31:16] | dim1[15:0]
  g1[3] = (int)((dim1 >> 16) | (tile0 << 16));               // dim1[31:16] | tile_dim0
  g1[4] = (int)tile1;                                        // tile_dim1 (tile_dim2=0)
  g1[5] = (int)stride0;                                      // tensor_dim0_stride[31:0]
  g1[6] = 0;
  g1[7] = 0;
  i32x4 g2 = {};
  i32x4 g3 = {};
#if __clang_major__ >= 23
  i32x8 g4 = {};
  __builtin_amdgcn_tensor_load_to_lds(g0, g1, g2, g3, g4, 0);
#else
  __builtin_amdgcn_tensor_load_to_lds(g0, g1, g2, g3, 0);
#endif
}
#endif

// ---------------------------------------------------------------------------
// fp32 -> f16 conversion (weights + activations, done once)
// ---------------------------------------------------------------------------
__global__ __launch_bounds__(256)
void ca_cvt_f16(const float* __restrict__ src, _Float16* __restrict__ dst, int n) {
  const int i = blockIdx.x * 256 + threadIdx.x;
  if (i < n) dst[i] = (_Float16)src[i];
}

// ---------------------------------------------------------------------------
// GEMM: Y[M,N](fp32) = X[M,K](f16) * W[N,K](f16)^T
// 128 threads (4 waves), 64x64 tile. A tile staged in LDS (pitch 34 halves);
// TDM double-buffered when available, manual h2 copies otherwise.
// ---------------------------------------------------------------------------
__device__ __forceinline__ void gemm_step(const _Float16* __restrict__ sap,
                                          const _Float16* __restrict__ wrow,
                                          int kk, int lane, int hi8, v8f acc[4]) {
  const _Float16* wb = wrow + kk + hi8;      // two 8-half runs (k hi8.., 16+hi8..), 16B aligned
  h8 blo = *(const h8*)wb;
  h8 bhi = *(const h8*)(wb + 16);
  v16h bf = __builtin_shufflevector(blo, bhi, 0, 1, 2, 3, 4, 5, 6, 7,
                                    8, 9, 10, 11, 12, 13, 14, 15);
#pragma unroll
  for (int mc = 0; mc < 4; ++mc) {
    v16h af;
    const int row = mc * 16 + (lane & 15);
#pragma unroll
    for (int j = 0; j < 8; ++j) {
      const int k0 = ((j & 3) << 1) + ((j >> 2) << 4) + hi8;
      h2 p = *(const h2*)&sap[row * 34 + k0];
      af[2 * j]     = p.x;
      af[2 * j + 1] = p.y;
    }
    acc[mc] = WMMA16x16x32(af, bf, acc[mc]);
  }
}

__global__ __launch_bounds__(128)
void ca_gemm_xwt(const _Float16* __restrict__ X, const _Float16* __restrict__ W,
                 float* __restrict__ Y, int M, int N, int K) {
  __shared__ _Float16 sa[2][64 * 34];
  const int tid    = threadIdx.x;
  const int wave   = tid >> 5;
  const int lane   = tid & 31;
  const int tile_m = blockIdx.x * 64;
  const int tile_n = blockIdx.y * 64;
  const int hi8    = (lane & 16) >> 1;

  v8f acc[4] = {};
  const int ncol = tile_n + wave * 16 + (lane & 15);
  const _Float16* wrow = W + (size_t)ncol * K;

#if HAVE_TDM
  const _Float16* xbase = X + (size_t)tile_m * K;
  const unsigned int sa_addr[2] = {lds_lo32(&sa[0][0]), lds_lo32(&sa[1][0])};
  if (wave == 0)  // prologue: tile for kk=0 (32 halves/row = 16 DWORDs -> pad_int=3)
    tdm_load_tile16(sa_addr[0], xbase, (unsigned)K, 64u, 32u, 64u, (unsigned)K, 3u);
  int cur = 0;
  for (int kk = 0; kk < K; kk += 32) {
    __syncthreads();  // all waves done reading buffer we are about to overwrite
    if (wave == 0) {
      if (kk + 32 < K) {
        tdm_load_tile16(sa_addr[cur ^ 1], xbase + kk + 32, (unsigned)K, 64u, 32u, 64u,
                        (unsigned)K, 3u);
        __builtin_amdgcn_s_wait_tensorcnt(1);  // current tile landed; next may fly
      } else {
        __builtin_amdgcn_s_wait_tensorcnt(0);
      }
    }
    __syncthreads();  // current tile visible to all waves
    gemm_step(sa[cur], wrow, kk, lane, hi8, acc);
    cur ^= 1;
  }
#else
  for (int kk = 0; kk < K; kk += 32) {
    __syncthreads();
    {  // manual staging: thread copies 16 contiguous halves
      const int r  = tid >> 1;
      const int c0 = (tid & 1) * 16;
      const _Float16* src = X + (size_t)(tile_m + r) * K + kk + c0;
      _Float16* dst = &sa[0][r * 34 + c0];
#pragma unroll
      for (int i = 0; i < 16; i += 2) *(h2*)&dst[i] = *(const h2*)&src[i];
    }
    __syncthreads();
    gemm_step(sa[0], wrow, kk, lane, hi8, acc);
  }
#endif

#pragma unroll
  for (int mc = 0; mc < 4; ++mc)
#pragma unroll
    for (int r = 0; r < 8; ++r) {
      const int grow = tile_m + mc * 16 + r + hi8;
      Y[(size_t)grow * N + ncol] = acc[mc][r];
    }
}

// ---------------------------------------------------------------------------
// Q path: RMSNorm(head) + RoPE, fp32 [B,L,768] -> f16 [B,H,L,64]
// One wave per (b,h,l); lane t owns dims t and t+32 (rotate_half pairing).
// ---------------------------------------------------------------------------
__global__ __launch_bounds__(256)
void ca_qnorm_rope(const float* __restrict__ qp, const float* __restrict__ cosv,
                   const float* __restrict__ sinv, const float* __restrict__ nw,
                   _Float16* __restrict__ q_h) {
  const int row  = blockIdx.x * 8 + (threadIdx.x >> 5);
  const int lane = threadIdx.x & 31;
  const int b = row / (NH * LQ);
  const int h = (row >> 11) % NH;
  const int l = row & (LQ - 1);

  const float* src = qp + ((size_t)(b * LQ + l)) * DIM + h * HD;
  float x1 = src[lane], x2 = src[lane + 32];
  float ss = x1 * x1 + x2 * x2;
#pragma unroll
  for (int m = 1; m < 32; m <<= 1) ss += __shfl_xor(ss, m, 32);
  const float rn = rsqrtf(ss * (1.0f / 64.0f) + 1e-6f);
  x1 *= rn * nw[lane];
  x2 *= rn * nw[lane + 32];

  const size_t cb = ((size_t)(b * LQ + l)) * HD;
  const float o1 = x1 * cosv[cb + lane]      - x2 * sinv[cb + lane];
  const float o2 = x2 * cosv[cb + lane + 32] + x1 * sinv[cb + lane + 32];

  _Float16* dst = q_h + ((size_t)((b * NH + h) * LQ + l)) * HD;
  dst[lane]      = (_Float16)o1;
  dst[lane + 32] = (_Float16)o2;
}

// ---------------------------------------------------------------------------
// KV path: K gets RMSNorm + RoPE; V converts. fp32 [B,L,1536] -> f16 pair.
// ---------------------------------------------------------------------------
__global__ __launch_bounds__(256)
void ca_kvnorm_rope(const float* __restrict__ kvp, const float* __restrict__ cosv,
                    const float* __restrict__ sinv, const float* __restrict__ nw,
                    _Float16* __restrict__ k_h, _Float16* __restrict__ v_h) {
  const int row  = blockIdx.x * 8 + (threadIdx.x >> 5);
  const int lane = threadIdx.x & 31;
  const int b = row / (NH * LK);
  const int h = (row >> 11) % NH;
  const int l = row & (LK - 1);

  const float* kb = kvp + ((size_t)(b * LK + l)) * (2 * DIM) + h * HD;
  const float* vb = kb + DIM;

  float x1 = kb[lane], x2 = kb[lane + 32];
  float ss = x1 * x1 + x2 * x2;
#pragma unroll
  for (int m = 1; m < 32; m <<= 1) ss += __shfl_xor(ss, m, 32);
  const float rn = rsqrtf(ss * (1.0f / 64.0f) + 1e-6f);
  x1 *= rn * nw[lane];
  x2 *= rn * nw[lane + 32];

  const size_t cb = ((size_t)(b * LK + l)) * HD;
  const float o1 = x1 * cosv[cb + lane]      - x2 * sinv[cb + lane];
  const float o2 = x2 * cosv[cb + lane + 32] + x1 * sinv[cb + lane + 32];

  const size_t dst = ((size_t)((b * NH + h) * LK + l)) * HD;
  k_h[dst + lane]      = (_Float16)o1;
  k_h[dst + lane + 32] = (_Float16)o2;
  v_h[dst + lane]      = vb[lane];
  v_h[dst + lane + 32] = vb[lane + 32];
}

// ---------------------------------------------------------------------------
// Flash attention: grid (LQ/64, B*NH), 128 threads (4 waves).
// K tile TDM-loaded into padded LDS; V hand-transposed; online softmax;
// P goes C-layout -> LDS -> A-layout; output written as f16 [B,Lq,768].
// ---------------------------------------------------------------------------
__global__ __launch_bounds__(128)
void ca_flash_attn(const _Float16* __restrict__ q_h, const _Float16* __restrict__ k_h,
                   const _Float16* __restrict__ v_h, _Float16* __restrict__ out) {
  __shared__ _Float16 kt[32 * 66];      // [key][d], pitch 66 (TDM pad or manual)
  __shared__ _Float16 vt[64 * 34];      // [d][key], pitch 34 (transposed)
  __shared__ _Float16 pb[4][16 * 34];   // per-wave P scratch [row][key]

  const int tid    = threadIdx.x;
  const int wave   = tid >> 5;
  const int lane   = tid & 31;
  const int bh     = blockIdx.y;        // b*12 + h
  const int tile_q = blockIdx.x * 64;
  const int hi8    = (lane & 16) >> 1;

  const size_t head_off = (size_t)bh * LK * HD;
  const _Float16* qp = q_h + head_off;
  const _Float16* kp = k_h + head_off;
  const _Float16* vp = v_h + head_off;

  // Q fragments: A-layout per half-wave = two contiguous h8 runs
  v16h aq[2];
  {
    const int qrow = tile_q + wave * 16 + (lane & 15);
    const _Float16* qr = qp + (size_t)qrow * HD;
#pragma unroll
    for (int f = 0; f < 2; ++f) {
      h8 lo = *(const h8*)&qr[f * 32 + hi8];
      h8 hi = *(const h8*)&qr[f * 32 + 16 + hi8];
      aq[f] = __builtin_shufflevector(lo, hi, 0, 1, 2, 3, 4, 5, 6, 7,
                                      8, 9, 10, 11, 12, 13, 14, 15);
    }
  }

  v8f o[4] = {};
  float mrow[8], lrow[8];
#pragma unroll
  for (int r = 0; r < 8; ++r) { mrow[r] = -1e30f; lrow[r] = 0.0f; }

  const int stage_r  = tid >> 2;        // 0..31 (key within tile)
  const int stage_c0 = (tid & 3) * 16;  // 0,16,32,48 (d)
#if HAVE_TDM
  const unsigned int kt_addr = lds_lo32(&kt[0]);
#endif

  for (int kt0 = 0; kt0 < LK; kt0 += 32) {
    __syncthreads();  // previous tile fully consumed
#if HAVE_TDM
    if (wave == 0)  // 64 halves/row = 32 DWORDs -> pad_int=4 -> pitch 66
      tdm_load_tile16(kt_addr, kp + (size_t)kt0 * HD, 64u, 32u, 64u, 32u, 64u, 4u);
#endif
    {  // stage V transposed [d][key] (and K manually if no TDM)
      const _Float16* sv = vp + (size_t)(kt0 + stage_r) * HD + stage_c0;
#if !HAVE_TDM
      const _Float16* sk = kp + (size_t)(kt0 + stage_r) * HD + stage_c0;
      _Float16* dk = &kt[stage_r * 66 + stage_c0];
#pragma unroll
      for (int i = 0; i < 16; i += 2) *(h2*)&dk[i] = *(const h2*)&sk[i];
#endif
#pragma unroll
      for (int i = 0; i < 16; ++i) vt[(stage_c0 + i) * 34 + stage_r] = sv[i];
      if (kt0 + 32 < LK)
        __builtin_prefetch(vp + (size_t)(kt0 + 32 + stage_r) * HD + stage_c0, 0, 0);
    }
#if HAVE_TDM
    if (wave == 0) __builtin_amdgcn_s_wait_tensorcnt(0);
#endif
    __syncthreads();

    // scores: 16q x 32k = two 16x16 tiles, each accumulated over 2 K-chunks
    v8f sc0 = {}, sc1 = {};
#pragma unroll
    for (int kc = 0; kc < 2; ++kc) {
      v16h bk0, bk1;
      const int key0 = lane & 15;
#pragma unroll
      for (int j = 0; j < 8; ++j) {
        const int d0 = kc * 32 + ((j & 3) << 1) + ((j >> 2) << 4) + hi8;
        h2 p0 = *(const h2*)&kt[key0 * 66 + d0];
        h2 p1 = *(const h2*)&kt[(key0 + 16) * 66 + d0];
        bk0[2 * j] = p0.x; bk0[2 * j + 1] = p0.y;
        bk1[2 * j] = p1.x; bk1[2 * j + 1] = p1.y;
      }
      sc0 = WMMA16x16x32(aq[kc], bk0, sc0);
      sc1 = WMMA16x16x32(aq[kc], bk1, sc1);
    }

    // online softmax (rows live in 16-lane half-wave groups)
    float alpha[8];
#pragma unroll
    for (int r = 0; r < 8; ++r) {
      const float s0 = sc0[r] * 0.125f;
      const float s1 = sc1[r] * 0.125f;
      float tm = fmaxf(s0, s1);
#pragma unroll
      for (int msk = 1; msk < 16; msk <<= 1) tm = fmaxf(tm, __shfl_xor(tm, msk, 32));
      const float mn = fmaxf(mrow[r], tm);
      const float al = __expf(mrow[r] - mn);
      const float p0 = __expf(s0 - mn);
      const float p1 = __expf(s1 - mn);
      float rs = p0 + p1;
#pragma unroll
      for (int msk = 1; msk < 16; msk <<= 1) rs += __shfl_xor(rs, msk, 32);
      lrow[r] = lrow[r] * al + rs;
      mrow[r] = mn;
      alpha[r] = al;
      const int prow = r + hi8;
      const int col  = lane & 15;
      pb[wave][prow * 34 + col]      = (_Float16)p0;
      pb[wave][prow * 34 + col + 16] = (_Float16)p1;
    }
#pragma unroll
    for (int t = 0; t < 4; ++t)
#pragma unroll
      for (int r = 0; r < 8; ++r) o[t][r] *= alpha[r];

    // per-wave LDS P round-trip: DS is in-order within a wave; fence compiler+hw
    asm volatile("s_wait_dscnt 0" ::: "memory");

    // P as A fragment (16x32)
    v16h pf;
    {
      const int prow = lane & 15;
#pragma unroll
      for (int j = 0; j < 8; ++j) {
        const int k0 = ((j & 3) << 1) + ((j >> 2) << 4) + hi8;
        h2 p = *(const h2*)&pb[wave][prow * 34 + k0];
        pf[2 * j]     = p.x;
        pf[2 * j + 1] = p.y;
      }
    }

    // O(16x64) += P(16x32) @ V(32x64)
#pragma unroll
    for (int t = 0; t < 4; ++t) {
      v16h bv;
      const int dg = t * 16 + (lane & 15);
#pragma unroll
      for (int j = 0; j < 8; ++j) {
        const int kk0 = ((j & 3) << 1) + ((j >> 2) << 4) + hi8;
        h2 p = *(const h2*)&vt[dg * 34 + kk0];
        bv[2 * j]     = p.x;
        bv[2 * j + 1] = p.y;
      }
      o[t] = WMMA16x16x32(pf, bv, o[t]);
    }
  }

  // epilogue: normalize, store f16 [B, Lq, H*HD]
  const int b = bh / NH;
  const int h = bh % NH;
#pragma unroll
  for (int r = 0; r < 8; ++r) {
    const float inv = 1.0f / lrow[r];
    const int qg = tile_q + wave * 16 + r + hi8;
    _Float16* orow = out + ((size_t)(b * LQ + qg)) * DIM + h * HD;
#pragma unroll
    for (int t = 0; t < 4; ++t) orow[t * 16 + (lane & 15)] = (_Float16)(o[t][r] * inv);
  }
}

// ---------------------------------------------------------------------------
extern "C" void kernel_launch(void* const* d_in, const int* in_sizes, int n_in,
                              void* d_out, int out_size, void* d_ws, size_t ws_size,
                              hipStream_t stream) {
  const float* queries  = (const float*)d_in[0];
  const float* context  = (const float*)d_in[1];
  const float* q_cos    = (const float*)d_in[2];
  const float* q_sin    = (const float*)d_in[3];
  const float* kv_cos   = (const float*)d_in[4];
  const float* kv_sin   = (const float*)d_in[5];
  const float* Wq       = (const float*)d_in[6];
  const float* Wkv      = (const float*)d_in[7];
  const float* Wo       = (const float*)d_in[8];
  const float* q_norm_w = (const float*)d_in[9];
  const float* k_norm_w = (const float*)d_in[10];
  float* out = (float*)d_out;

  const size_t rows = (size_t)BATCH * LQ;  // 8192
  char* ws = (char*)d_ws;
  size_t off = 0;
  _Float16* queries_h = (_Float16*)(ws + off); off += rows * DIM * sizeof(_Float16);
  _Float16* context_h = (_Float16*)(ws + off); off += rows * DIM * sizeof(_Float16);
  _Float16* Wq_h  = (_Float16*)(ws + off); off += (size_t)DIM * DIM * sizeof(_Float16);
  _Float16* Wkv_h = (_Float16*)(ws + off); off += (size_t)2 * DIM * DIM * sizeof(_Float16);
  _Float16* Wo_h  = (_Float16*)(ws + off); off += (size_t)DIM * DIM * sizeof(_Float16);
  float* q_proj   = (float*)(ws + off); off += rows * DIM * sizeof(float);
  float* kv_proj  = (float*)(ws + off); off += rows * (2 * DIM) * sizeof(float);
  _Float16* q_h   = (_Float16*)(ws + off); off += rows * DIM * sizeof(_Float16);
  _Float16* k_h   = (_Float16*)(ws + off); off += rows * DIM * sizeof(_Float16);
  _Float16* v_h   = (_Float16*)(ws + off); off += rows * DIM * sizeof(_Float16);
  _Float16* attn_h = queries_h;  // queries_h dead after the Q projection; alias it

  // 0) one-time fp32 -> f16 conversions
  const int n_act = (int)(rows * DIM);
  ca_cvt_f16<<<(n_act + 255) / 256, 256, 0, stream>>>(queries, queries_h, n_act);
  ca_cvt_f16<<<(n_act + 255) / 256, 256, 0, stream>>>(context, context_h, n_act);
  ca_cvt_f16<<<(DIM * DIM + 255) / 256, 256, 0, stream>>>(Wq, Wq_h, DIM * DIM);
  ca_cvt_f16<<<(2 * DIM * DIM + 255) / 256, 256, 0, stream>>>(Wkv, Wkv_h, 2 * DIM * DIM);
  ca_cvt_f16<<<(DIM * DIM + 255) / 256, 256, 0, stream>>>(Wo, Wo_h, DIM * DIM);

  // 1) projections (f16 WMMA, fp32 out for accurate RMSNorm)
  ca_gemm_xwt<<<dim3(rows / 64, DIM / 64), 128, 0, stream>>>(
      queries_h, Wq_h, q_proj, (int)rows, DIM, DIM);
  ca_gemm_xwt<<<dim3(rows / 64, (2 * DIM) / 64), 128, 0, stream>>>(
      context_h, Wkv_h, kv_proj, (int)rows, 2 * DIM, DIM);

  // 2) per-head RMSNorm + RoPE -> f16 [B,H,L,HD]
  const int norm_blocks = (BATCH * NH * LQ) / 8;
  ca_qnorm_rope<<<norm_blocks, 256, 0, stream>>>(q_proj, q_cos, q_sin, q_norm_w, q_h);
  ca_kvnorm_rope<<<norm_blocks, 256, 0, stream>>>(kv_proj, kv_cos, kv_sin, k_norm_w, k_h, v_h);

  // 3) flash attention -> f16 [B,Lq,768]
  ca_flash_attn<<<dim3(LQ / 64, BATCH * NH), 128, 0, stream>>>(q_h, k_h, v_h, attn_h);

  // 4) output projection -> fp32 d_out
  ca_gemm_xwt<<<dim3(rows / 64, DIM / 64), 128, 0, stream>>>(
      attn_h, Wo_h, out, (int)rows, DIM, DIM);
}
